// MultiHeadAttention_25460566130809
// MI455X (gfx1250) — matrix operand order repaired and checked by reading, not compiled, not run on hardware
//
#include <hip/hip_runtime.h>
#include <hip/hip_bf16.h>

// ---------------------------------------------------------------------------
// MHA forward for MI455X (gfx1250): bf16 WMMA + TDM tile staging.
// B=2, S=2048, E=1024, H=16, D=64.
// Pipeline: convert x->bf16, transpose-convert W->bf16 [N][K],
//           TDM-staged bf16 GEMMs (QKV + out-proj), flash attention.
// ---------------------------------------------------------------------------

typedef __attribute__((ext_vector_type(16))) __bf16 v16bf;
typedef __attribute__((ext_vector_type(8)))  __bf16 v8bf;
typedef __attribute__((ext_vector_type(8)))  float  v8f;
typedef __attribute__((ext_vector_type(4)))  unsigned u32x4;
typedef __attribute__((ext_vector_type(8)))  int      i32x8;
typedef __attribute__((ext_vector_type(4)))  int      i32x4;

#define BD   2
#define SD   2048
#define ED   1024
#define HD   16
#define DD   64
#define NTOK 4096

__device__ __forceinline__ unsigned short f2bf(float f) {
    unsigned u = __builtin_bit_cast(unsigned, f);
    unsigned r = u + 0x7fffu + ((u >> 16) & 1u);   // round-to-nearest-even
    return (unsigned short)(r >> 16);
}

union FragA { v16bf v; v8bf h[2]; };

// ---------------------------------------------------------------------------
// TDM: 2D tile load Global -> LDS, data_size = 2B. D# per ISA 08 §8.
// This toolchain exposes the 6-arg builtin: (g0, g1, g2, g3, g4, cpol).
// ---------------------------------------------------------------------------
#if __has_builtin(__builtin_amdgcn_tensor_load_to_lds)
#define HAVE_TDM 1
__device__ __forceinline__ void tdm_load_2d(unsigned lds_off, const void* gptr,
                                            unsigned tensor_d0, unsigned tensor_d1,
                                            unsigned tile_d0, unsigned tile_d1,
                                            unsigned stride0)
{
    unsigned long long ga = (unsigned long long)(uintptr_t)gptr;
    u32x4 g0;
    g0.x = 1u;                                   // count=1, user descriptor
    g0.y = lds_off;                              // lds_addr [63:32]
    g0.z = (unsigned)ga;                         // global_addr [95:64]
    g0.w = (unsigned)((ga >> 32) & 0x01ffffffu)  // global_addr [120:96]
         | 0x80000000u;                          // type=2 ("image") [127:126]
    i32x8 g1;
    g1[0] = (int)(1u << 16);                                     // data_size=2B
    g1[1] = (int)((tensor_d0 & 0xffffu) << 16);                  // dim0 lo16
    g1[2] = (int)(((tensor_d0 >> 16) & 0xffffu)
                | ((tensor_d1 & 0xffffu) << 16));                // dim0 hi/dim1 lo
    g1[3] = (int)(((tensor_d1 >> 16) & 0xffffu)
                | (tile_d0 << 16));                              // dim1 hi/tile0
    g1[4] = (int)(tile_d1 & 0xffffu);                            // tile1, tile2=0
    g1[5] = (int)stride0;                                        // dim0 stride lo32
    g1[6] = 0;                                                   // stride hi, dim1stride
    g1[7] = 0;
    i32x4 z4 = {0, 0, 0, 0};
    i32x8 z8 = {0, 0, 0, 0, 0, 0, 0, 0};
    __builtin_amdgcn_tensor_load_to_lds(g0, g1, z4, z4, z8, 0);
}
#else
#define HAVE_TDM 0
#endif

// ---------------------------------------------------------------------------
// fp32 -> bf16 bulk convert (x). 8 elems / thread.
// ---------------------------------------------------------------------------
__global__ __launch_bounds__(256) void conv_bf16(const float* __restrict__ in,
                                                 unsigned short* __restrict__ out)
{
    size_t i = ((size_t)blockIdx.x * 256 + threadIdx.x) * 8;
    float4 a = *(const float4*)(in + i);
    float4 b = *(const float4*)(in + i + 4);
    uint4 o;
    o.x = (unsigned)f2bf(a.x) | ((unsigned)f2bf(a.y) << 16);
    o.y = (unsigned)f2bf(a.z) | ((unsigned)f2bf(a.w) << 16);
    o.z = (unsigned)f2bf(b.x) | ((unsigned)f2bf(b.y) << 16);
    o.w = (unsigned)f2bf(b.z) | ((unsigned)f2bf(b.w) << 16);
    *(uint4*)(out + i) = o;
}

// ---------------------------------------------------------------------------
// W [K][N] fp32 -> Wt [N][K] bf16, 64x64 tiles through LDS.
// ---------------------------------------------------------------------------
__global__ __launch_bounds__(256) void transpose_bf16(const float* __restrict__ in,
                                                      unsigned short* __restrict__ out)
{
    __shared__ __align__(16) unsigned short t[64][72];   // 144B rows, 16B aligned
    const int tid = threadIdx.x;
    const int kb = blockIdx.x * 64;   // K base
    const int nb = blockIdx.y * 64;   // N base
#pragma unroll
    for (int it = 0; it < 4; ++it) {
        int idx = tid + it * 256;
        int kr = idx >> 4, c4 = (idx & 15) << 2;
        const float4 f = *(const float4*)(in + (size_t)(kb + kr) * ED + nb + c4);
        t[c4 + 0][kr] = f2bf(f.x);
        t[c4 + 1][kr] = f2bf(f.y);
        t[c4 + 2][kr] = f2bf(f.z);
        t[c4 + 3][kr] = f2bf(f.w);
    }
    __syncthreads();
#pragma unroll
    for (int it = 0; it < 2; ++it) {
        int idx = tid + it * 256;
        int n = idx >> 3, k8 = (idx & 7) << 3;
        *(uint4*)(out + (size_t)(nb + n) * ED + kb + k8) = *(const uint4*)&t[n][k8];
    }
}

// ---------------------------------------------------------------------------
// GEMM: A bf16 [4096][1024] @ Wt bf16 [1024 n][1024 k] (pre-transposed).
// mode 0: Q (bf16 [B][H][S][D], *1/8)   mode 1: K (bf16 [B][H][S][D])
// mode 2: V (bf16 [B][H][D][S])         mode 3: f32 out [N][E]
// Tile M=128,N=64,K=64; 8 waves (4x2), wave = 32x32; TDM-staged LDS tiles.
// ---------------------------------------------------------------------------
__global__ __launch_bounds__(256) void mha_gemm_bf16(
    const unsigned short* __restrict__ A, const unsigned short* __restrict__ Wt,
    const float* __restrict__ bias, void* __restrict__ outp, int mode)
{
    __shared__ __align__(32) unsigned short As[128 * 64];   // 16 KB
    __shared__ __align__(32) unsigned short Ws[64 * 64];    //  8 KB

    const int tid  = threadIdx.x;
    const int lane = tid & 31;
    const int wv   = tid >> 5;
    const int wm   = wv & 3;
    const int wn   = wv >> 2;
    const int hf   = lane >> 4;
    const int lm   = lane & 15;
    const int Mblk = blockIdx.x * 128;
    const int Nblk = blockIdx.y * 64;

    const v8f vz = {0.f, 0.f, 0.f, 0.f, 0.f, 0.f, 0.f, 0.f};
    v8f acc[2][2];
    acc[0][0] = vz; acc[0][1] = vz; acc[1][0] = vz; acc[1][1] = vz;

    for (int kb = 0; kb < ED; kb += 64) {
#if HAVE_TDM
        if (wv == 0) {
            tdm_load_2d((unsigned)(uintptr_t)(void*)As,
                        A + (size_t)Mblk * ED + kb, ED, NTOK, 64, 128, ED);
            tdm_load_2d((unsigned)(uintptr_t)(void*)Ws,
                        Wt + (size_t)Nblk * ED + kb, ED, ED, 64, 64, ED);
            __builtin_amdgcn_s_wait_tensorcnt(0);
        }
#else
#pragma unroll
        for (int it = 0; it < 4; ++it) {          // A tile: 1024 x uint4
            int idx = tid + it * 256;
            int r = idx >> 3, c8 = (idx & 7) << 3;
            *(uint4*)&As[r * 64 + c8] =
                *(const uint4*)(A + (size_t)(Mblk + r) * ED + kb + c8);
        }
#pragma unroll
        for (int it = 0; it < 2; ++it) {          // W tile: 512 x uint4
            int idx = tid + it * 256;
            int n = idx >> 3, c8 = (idx & 7) << 3;
            *(uint4*)&Ws[n * 64 + c8] =
                *(const uint4*)(Wt + (size_t)(Nblk + n) * ED + kb + c8);
        }
#endif
        __syncthreads();

#pragma unroll
        for (int kh = 0; kh < 2; ++kh) {
            FragA af[2];
            v16bf bf[2];
#pragma unroll
            for (int mt = 0; mt < 2; ++mt) {
                int m = wm * 32 + mt * 16 + lm;
                af[mt].h[0] = *(const v8bf*)&As[m * 64 + kh * 32 + hf * 16];
                af[mt].h[1] = *(const v8bf*)&As[m * 64 + kh * 32 + hf * 16 + 8];
            }
#pragma unroll
            for (int nt = 0; nt < 2; ++nt) {
                int n = wn * 32 + nt * 16 + lm;
                bf[nt] = *(const v16bf*)&Ws[n * 64 + kh * 32 + hf * 16];
            }
#pragma unroll
            for (int mt = 0; mt < 2; ++mt)
#pragma unroll
                for (int nt = 0; nt < 2; ++nt)
                    acc[mt][nt] = __builtin_amdgcn_wmma_f32_16x16x32_bf16(
                        false, af[mt].v, false, bf[nt], (short)0, acc[mt][nt],
                        false, false);
        }
        __syncthreads();
    }

    // ---- epilogue ----
#pragma unroll
    for (int mt = 0; mt < 2; ++mt) {
#pragma unroll
        for (int nt = 0; nt < 2; ++nt) {
            int row0 = Mblk + wm * 32 + mt * 16 + hf * 8;
            int col  = Nblk + wn * 32 + nt * 16 + lm;
            float bc = bias[col];
            if (mode == 3) {
                float* O = (float*)outp;
#pragma unroll
                for (int i = 0; i < 8; ++i)
                    O[(size_t)(row0 + i) * ED + col] = acc[mt][nt][i] + bc;
            } else {
                unsigned short* O = (unsigned short*)outp;
                int hh = col >> 6, dd = col & 63;
#pragma unroll
                for (int i = 0; i < 8; ++i) {
                    int r = row0 + i;
                    int b = r >> 11, s = r & (SD - 1);
                    float v = acc[mt][nt][i] + bc;
                    if (mode == 0) v *= 0.125f;           // 1/sqrt(64) into Q
                    size_t off;
                    if (mode == 2)                        // V: [b][h][d][s]
                        off = (((size_t)(b * HD + hh) * DD + dd) * SD + s);
                    else                                  // Q/K: [b][h][s][d]
                        off = (((size_t)(b * HD + hh) * SD + s) * DD + dd);
                    O[off] = f2bf(v);
                }
            }
        }
    }
}

// ---------------------------------------------------------------------------
// Flash attention: block = (64 q-rows, head, batch); 4 waves x 16 q-rows.
// 32-key chunks: 4 WMMAs scores + online softmax + 4 WMMAs P@V.
// Output written directly in bf16 (feeds out-proj GEMM).
// ---------------------------------------------------------------------------
__global__ __launch_bounds__(128) void mha_flash_attn(
    const unsigned short* __restrict__ Q, const unsigned short* __restrict__ K,
    const unsigned short* __restrict__ Vt, const int* __restrict__ mask,
    unsigned short* __restrict__ Oout)
{
    __shared__ __align__(32) unsigned short Pl[4][16 * 40];

    const int tid  = threadIdx.x;
    const int lane = tid & 31;
    const int wv   = tid >> 5;
    const int hf   = lane >> 4;
    const int lm   = lane & 15;
    const int qt = blockIdx.x, hh = blockIdx.y, bb = blockIdx.z;
    const int qbase = qt * 64 + wv * 16;
    const size_t bh = (size_t)(bb * HD + hh);

    const unsigned short* Qp = Q  + (bh * SD + qbase) * DD;
    const unsigned short* Kp = K  + bh * SD * DD;
    const unsigned short* Vp = Vt + bh * DD * SD;
    const int* mp = mask + bb * SD;

    FragA qa0, qa1;
    {
        const unsigned short* qr = Qp + (size_t)lm * DD;
        qa0.h[0] = *(const v8bf*)(qr + hf * 16);
        qa0.h[1] = *(const v8bf*)(qr + hf * 16 + 8);
        qa1.h[0] = *(const v8bf*)(qr + 32 + hf * 16);
        qa1.h[1] = *(const v8bf*)(qr + 32 + hf * 16 + 8);
    }

    const v8f vz = {0.f, 0.f, 0.f, 0.f, 0.f, 0.f, 0.f, 0.f};
    v8f oacc[4];
    oacc[0] = vz; oacc[1] = vz; oacc[2] = vz; oacc[3] = vz;
    float mrow[8], lrow[8];
#pragma unroll
    for (int i = 0; i < 8; ++i) { mrow[i] = -INFINITY; lrow[i] = 0.f; }

    unsigned short* pl = &Pl[wv][0];

    for (int kb = 0; kb < SD; kb += 32) {
        if (kb + 32 < SD) {   // global_prefetch_b8 next K/V rows
            __builtin_prefetch(Kp + (size_t)(kb + 32 + lm) * DD, 0, 1);
            __builtin_prefetch(Vp + (size_t)(lm)*SD + kb + 32, 0, 1);
        }
        v8f c[2];
#pragma unroll
        for (int sub = 0; sub < 2; ++sub) {
            const unsigned short* kr = Kp + (size_t)(kb + sub * 16 + lm) * DD;
            v16bf kb0 = *(const v16bf*)(kr + hf * 16);
            v16bf kb1 = *(const v16bf*)(kr + 32 + hf * 16);
            v8f z = vz;
            z = __builtin_amdgcn_wmma_f32_16x16x32_bf16(false, qa0.v, false, kb0,
                                                        (short)0, z, false, false);
            z = __builtin_amdgcn_wmma_f32_16x16x32_bf16(false, qa1.v, false, kb1,
                                                        (short)0, z, false, false);
            int mv = mp[kb + sub * 16 + lm];
#pragma unroll
            for (int i = 0; i < 8; ++i)
                c[sub][i] = (mv == 0) ? INFINITY : z[i];   // mirror reference
        }
#pragma unroll
        for (int i = 0; i < 8; ++i) {
            float rm = fmaxf(c[0][i], c[1][i]);
#pragma unroll
            for (int ms = 1; ms < 16; ms <<= 1)
                rm = fmaxf(rm, __shfl_xor(rm, ms, 32));
            float mnew  = fmaxf(mrow[i], rm);
            float alpha = __expf(mrow[i] - mnew);
            float p0 = __expf(c[0][i] - mnew);
            float p1 = __expf(c[1][i] - mnew);
            float rs = p0 + p1;
#pragma unroll
            for (int ms = 1; ms < 16; ms <<= 1)
                rs += __shfl_xor(rs, ms, 32);
            lrow[i] = lrow[i] * alpha + rs;
            mrow[i] = mnew;
            c[0][i] = p0; c[1][i] = p1;
#pragma unroll
            for (int dt = 0; dt < 4; ++dt) oacc[dt][i] *= alpha;
        }
#pragma unroll
        for (int i = 0; i < 8; ++i) {
            pl[(hf * 8 + i) * 40 + lm]      = f2bf(c[0][i]);
            pl[(hf * 8 + i) * 40 + 16 + lm] = f2bf(c[1][i]);
        }
        asm volatile("s_wait_dscnt 0" ::: "memory");
        FragA pa;
        {
            const unsigned short* pr = pl + lm * 40;
            pa.h[0] = *(const v8bf*)(pr + hf * 16);
            pa.h[1] = *(const v8bf*)(pr + hf * 16 + 8);
        }
#pragma unroll
        for (int dt = 0; dt < 4; ++dt) {
            const unsigned short* vr = Vp + (size_t)(dt * 16 + lm) * SD + kb + hf * 16;
            v16bf vb = *(const v16bf*)vr;
            oacc[dt] = __builtin_amdgcn_wmma_f32_16x16x32_bf16(
                false, pa.v, false, vb, (short)0, oacc[dt], false, false);
        }
    }

#pragma unroll
    for (int i = 0; i < 8; ++i) {
        float inv = 1.0f / lrow[i];
        int srow = qbase + hf * 8 + i;
        size_t base = ((size_t)bb * SD + srow) * ED + hh * DD;
#pragma unroll
        for (int dt = 0; dt < 4; ++dt)
            Oout[base + dt * 16 + lm] = f2bf(oacc[dt][i] * inv);
    }
}

// ---------------------------------------------------------------------------
extern "C" void kernel_launch(void* const* d_in, const int* in_sizes, int n_in,
                              void* d_out, int out_size, void* d_ws, size_t ws_size,
                              hipStream_t stream)
{
    (void)in_sizes; (void)n_in; (void)out_size; (void)ws_size;
    const float* x  = (const float*)d_in[0];
    const int* mask = (const int*)d_in[1];
    const float* Wq = (const float*)d_in[2];
    const float* bq = (const float*)d_in[3];
    const float* Wk = (const float*)d_in[4];
    const float* bk = (const float*)d_in[5];
    const float* Wv = (const float*)d_in[6];
    const float* bv = (const float*)d_in[7];
    const float* Wo = (const float*)d_in[8];
    const float* bo = (const float*)d_in[9];
    float* out = (float*)d_out;

    unsigned short* ws = (unsigned short*)d_ws;
    unsigned short* xb    = ws;                                // 8 MB
    unsigned short* Wtq   = xb  + (size_t)NTOK * ED;           // 2 MB each
    unsigned short* Wtk   = Wtq + (size_t)ED * ED;
    unsigned short* Wtv   = Wtk + (size_t)ED * ED;
    unsigned short* Wto   = Wtv + (size_t)ED * ED;
    unsigned short* Qb    = Wto + (size_t)ED * ED;             // 8 MB each
    unsigned short* Kb    = Qb  + (size_t)NTOK * ED;
    unsigned short* Vb    = Kb  + (size_t)NTOK * ED;
    unsigned short* attnb = Vb  + (size_t)NTOK * ED;           // 8 MB

    conv_bf16<<<(NTOK * ED) / (256 * 8), 256, 0, stream>>>(x, xb);
    dim3 gt(ED / 64, ED / 64, 1);
    transpose_bf16<<<gt, 256, 0, stream>>>(Wq, Wtq);
    transpose_bf16<<<gt, 256, 0, stream>>>(Wk, Wtk);
    transpose_bf16<<<gt, 256, 0, stream>>>(Wv, Wtv);
    transpose_bf16<<<gt, 256, 0, stream>>>(Wo, Wto);

    dim3 gg(NTOK / 128, ED / 64, 1);
    mha_gemm_bf16<<<gg, 256, 0, stream>>>(xb, Wtq, bq, (void*)Qb, 0);
    mha_gemm_bf16<<<gg, 256, 0, stream>>>(xb, Wtk, bk, (void*)Kb, 1);
    mha_gemm_bf16<<<gg, 256, 0, stream>>>(xb, Wtv, bv, (void*)Vb, 2);

    mha_flash_attn<<<dim3(SD / 64, HD, BD), 128, 0, stream>>>(Qb, Kb, Vb, mask, attnb);

    mha_gemm_bf16<<<gg, 256, 0, stream>>>(attnb, Wto, bo, (void*)out, 3);
}
